// GATrBlock_61564061221554
// MI455X (gfx1250) — matrix operand order repaired and checked by reading, not compiled, not run on hardware
//
#include <hip/hip_runtime.h>
#include <math.h>

// ---------------------------------------------------------------------------
// GATr block for MI455X (gfx1250, wave32, WMMA 16x16x32 bf16).
// All GEMMs (equi_linear channel mixing, QK^T scores, A*V) run on
// v_wmma_f32_16x16x32_bf16 with f32 accumulation; softmax / norms / geometric
// bilinear stay f32 on the VALU. Everything is L2-resident (192MB), so a
// two-pass (scores->softmax->AV) attention using 128KB LDS score rows beats
// flash-style recomputation here.
//
// Component indices are template parameters so all algebra metadata constant-
// folds; wave dispatch goes through readfirstlane -> scalar branches, keeping
// EXEC all-1s around every WMMA as the ISA requires.
// ---------------------------------------------------------------------------

typedef __bf16 bf16;
typedef __bf16 v8bf  __attribute__((ext_vector_type(8)));
typedef __bf16 v16bf __attribute__((ext_vector_type(16)));
typedef float  v8f   __attribute__((ext_vector_type(8)));

#define DEV static __device__ __forceinline__

// ----------------------------- algebra tables ------------------------------
constexpr int kMasks[16] = {0,1,2,4,8, 3,5,6,9,10,12, 7,11,13,14, 15};

constexpr int popc4(int x){ int c=0; for(int i=0;i<4;++i) c += (x>>i)&1; return c; }
constexpr int rsgn(int a,int b){ int n=a>>1,t=0; while(n){ t+=popc4(n&b); n>>=1; } return (t&1)?-1:1; }

struct GeomTables {
  signed char gt[16][16]; signed char gs[16][16];   // geometric product
  signed char jt[16][16]; signed char js[16][16];   // join
};
constexpr GeomTables makeTables(){
  GeomTables T{};
  int inv[16] = {};
  for(int i=0;i<16;++i) inv[kMasks[i]] = i;
  for(int ia=0; ia<16; ++ia){
    for(int ib=0; ib<16; ++ib){
      int a = kMasks[ia], b = kMasks[ib];
      if(a & b & 1){ T.gt[ia][ib] = -1; T.gs[ia][ib] = 0; }
      else { T.gt[ia][ib] = (signed char)inv[a^b]; T.gs[ia][ib] = (signed char)rsgn(a,b); }
      int da = 15^a, db = 15^b;
      if(da & db){ T.jt[ia][ib] = -1; T.js[ia][ib] = 0; }
      else {
        int dc = da^db, res = 15^dc;
        int s = rsgn(a,da)*rsgn(b,db)*rsgn(da,db)*rsgn(dc,res);
        T.jt[ia][ib] = (signed char)inv[res]; T.js[ia][ib] = (signed char)s;
      }
    }
  }
  return T;
}
constexpr GeomTables TB = makeTables();

// Per-component equi_linear metadata (basis order above). Used only with
// compile-time indices -> fully constant-folded.
constexpr int GRADE[16]   = {0,1,1,1,1, 2,2,2,2,2,2, 3,3,3,3, 4};
constexpr bool HASE0[16]  = {0,1,0,0,0, 1,1,0,1,0,0, 1,1,1,0, 1};
constexpr int PARTNER[16] = {0,0,0,0,0, 2,3,0,4,0,0, 7,9,10,0, 14};
constexpr int PAIRB[16]   = {0,5,0,0,0, 6,6,0,6,0,0, 7,7,7,0, 8};
constexpr bool ISINNER[16]= {1,0,1,1,1, 0,0,1,0,1,1, 0,0,0,1, 0};
constexpr int QSLOT[16]   = {0,0,1,2,3, 0,0,4,0,5,6, 0,0,0,7, 0};

// ------------------------------- helpers -----------------------------------
DEV bf16  tob(float f){ return (bf16)f; }

DEV v16bf cat16(v8bf lo, v8bf hi){
  v16bf r;
#pragma unroll
  for(int i=0;i<8;++i){ r[i] = lo[i]; r[8+i] = hi[i]; }
  return r;
}
// A fragment (16x32, row-major source). p = base + m*ld + k0 + 8*kq.
DEV v16bf ldA(const bf16* p){
  v8bf lo = *(const v8bf*)(p);
  v8bf hi = *(const v8bf*)(p + 16);
  return cat16(lo, hi);
}
// B fragment (32x16, "n-major" source [n][k]). p = base + n*ld + k0 + 8*kq.
DEV v16bf ldB(const bf16* p){
  v8bf lo = *(const v8bf*)(p);
  v8bf hi = *(const v8bf*)(p + 16);
  return cat16(lo, hi);
}
DEV v8f wmma_bf(v16bf a, v16bf b, v8f c){
  return __builtin_amdgcn_wmma_f32_16x16x32_bf16(false, a, false, b, (short)0, c, false, false);
}
DEV v8f vzero(){ v8f z = {0.f,0.f,0.f,0.f,0.f,0.f,0.f,0.f}; return z; }
DEV float gelu_exact(float x){ return 0.5f*x*(1.0f + erff(x*0.70710678118654752f)); }
DEV int wave_id(){ return __builtin_amdgcn_readfirstlane((int)threadIdx.x) >> 5; }

// Uniform scalar 8-way dispatch (wv lives in an SGPR -> s_cmp/s_cbranch).
#define WAVE8(wv, FN) \
  { if((wv)==0){FN(0)} else if((wv)==1){FN(1)} else if((wv)==2){FN(2)} \
    else if((wv)==3){FN(3)} else if((wv)==4){FN(4)} else if((wv)==5){FN(5)} \
    else if((wv)==6){FN(6)} else {FN(7)} }

// equi_linear 16x16 output tile for compile-time component J.
// Input layout: [token][16 comps][64 ch] bf16 (inb offset by tokbase).
// Weights: Wp[b][OUT][64] bf16, WSB = OUT*64.
template<int J, int WSB>
DEV v8f lin_tile(const bf16* inb, const bf16* Wp, int o, int ln, int kq){
  v8f acc = vzero();
  const bf16* arow = inb + (size_t)ln*1024 + J*64 + 8*kq;
  const bf16* wb   = Wp + (size_t)(GRADE[J]*WSB) + (size_t)o*64 + 8*kq;
  acc = wmma_bf(ldA(arow),      ldB(wb),      acc);
  acc = wmma_bf(ldA(arow + 32), ldB(wb + 32), acc);
  if constexpr (HASE0[J]){
    const bf16* prow = inb + (size_t)ln*1024 + PARTNER[J]*64 + 8*kq;
    const bf16* wp   = Wp + (size_t)(PAIRB[J]*WSB) + (size_t)o*64 + 8*kq;
    acc = wmma_bf(ldA(prow),      ldB(wp),      acc);
    acc = wmma_bf(ldA(prow + 32), ldB(wp + 32), acc);
  }
  return acc;
}

// ----------------------------- K1: weight prep -----------------------------
// w[o][i][9] f32  ->  Wp[b][o][i] bf16 (B-fragment friendly: k=i contiguous)
__global__ void k_prep_w(const float* __restrict__ src, bf16* __restrict__ dst, int OUT){
  int i = blockIdx.x*256 + threadIdx.x;
  int total = OUT*64*9;
  if(i >= total) return;
  int k = i & 63; int rest = i >> 6; int o = rest % OUT; int b = rest / OUT;
  dst[i] = tob(src[(o*64 + k)*9 + b]);
}

// -------------- K2: equi_norm + equi_linear(w1) -> Q (bf16), V (bf16) ------
template<int J>
DEV void k2_comp(const bf16* xn, const bf16* __restrict__ Wp,
                 bf16* __restrict__ Qb, bf16* __restrict__ Vb,
                 int tokbase, int ln, int kq){
  const bf16* arow = xn + (size_t)(J*16 + ln)*64 + 8*kq;
  v16bf a0 = ldA(arow), a1 = ldA(arow + 32);
#pragma unroll
  for(int ct=0; ct<4; ++ct){
    int o = ct*16 + ln;
    v8f acc = vzero();
    const bf16* wb = Wp + (size_t)(GRADE[J]*4096) + (size_t)o*64 + 8*kq;
    acc = wmma_bf(a0, ldB(wb),      acc);
    acc = wmma_bf(a1, ldB(wb + 32), acc);
    if constexpr (HASE0[J]){
      const bf16* prow = xn + (size_t)(PARTNER[J]*16 + ln)*64 + 8*kq;
      const bf16* wp2  = Wp + (size_t)(PAIRB[J]*4096) + (size_t)o*64 + 8*kq;
      acc = wmma_bf(ldA(prow),      ldB(wp2),      acc);
      acc = wmma_bf(ldA(prow + 32), ldB(wp2 + 32), acc);
    }
    int t0 = tokbase + 8*kq;
    int batch = t0 >> 11, tin = t0 & 2047;
    v8bf vb;
#pragma unroll
    for(int v=0; v<8; ++v) vb[v] = tob(acc[v]);
    *(v8bf*)(Vb + ((size_t)batch*1024 + J*64 + o)*2048 + tin) = vb;  // contiguous
    if constexpr (ISINNER[J]){
#pragma unroll
      for(int v=0; v<8; ++v) Qb[(size_t)(t0+v)*512 + QSLOT[J]*64 + o] = vb[v];
    }
  }
}

__global__ __launch_bounds__(256) void k_norm_lin1(const float* __restrict__ x,
                                                   const bf16* __restrict__ Wp,
                                                   bf16* __restrict__ Qb,
                                                   bf16* __restrict__ Vb){
  extern __shared__ char smem[];
  float* ldsX = (float*)smem;                       // [16tok][64ch][16comp] f32
  bf16*  xn   = (bf16*)(smem + 65536);              // [16comp][16tok][64ch] normed bf16
  float* ssum = (float*)(smem + 65536 + 32768);     // [16]
  int tid = threadIdx.x;
  int tokbase = blockIdx.x * 16;
  if(tid < 16) ssum[tid] = 0.f;
  __syncthreads();
  const float* xg = x + (size_t)tokbase*1024;
  for(int i=tid; i<16384; i+=256){
    float v = xg[i];
    ldsX[i] = v;
    // inner components = even masks: idx 0,2,3,4,7,9,10,14
    int comp = i & 15;
    bool inner = (comp==0)|(comp==2)|(comp==3)|(comp==4)|(comp==7)|(comp==9)|(comp==10)|(comp==14);
    if(inner) atomicAdd(&ssum[i >> 10], v*v);
  }
  __syncthreads();
  if(tid < 16) ssum[tid] = rsqrtf(ssum[tid]*(1.0f/64.0f) + 1e-6f);
  __syncthreads();
  for(int i=tid; i<16384; i+=256){
    int c = i & 63; int m = (i>>6) & 15; int j = i >> 10;
    xn[i] = tob(ldsX[(m*64 + c)*16 + j] * ssum[m]);
  }
  __syncthreads();

  int lane = tid & 31, ln = lane & 15, kq = lane >> 4;
  int wv = wave_id();
#define K2W(W) { k2_comp<2*W>(xn, Wp, Qb, Vb, tokbase, ln, kq); \
                 k2_comp<2*W+1>(xn, Wp, Qb, Vb, tokbase, ln, kq); }
  WAVE8(wv, K2W)
#undef K2W
}

// -------------------- K3: scores (QQ^T) + softmax -> A bf16 ----------------
__global__ __launch_bounds__(128) void k_scores(const bf16* __restrict__ Qb,
                                                bf16* __restrict__ A){
  extern __shared__ char smem[];
  float* sc   = (float*)smem;                 // [16][2048] f32 = 128KB
  float* pmax = (float*)(smem + 131072);      // [16][8]
  float* psum = pmax + 128;                   // [16][8]
  float* rmax = psum + 128;                   // [16]
  float* rsum = rmax + 16;                    // [16]
  int tid = threadIdx.x, lane = tid & 31;
  int ln = lane & 15, kq = lane >> 4;
  int wv = wave_id();
  int wg = blockIdx.x;
  int batch = wg >> 7, qt = wg & 127;
  int qg = batch*2048 + qt*16;
  const float scale = 0.044194173824159216f;  // 1/sqrt(8*64)

  const bf16* qrow = Qb + (size_t)(qg + ln)*512 + 8*kq;
  for(int kt = wv; kt < 128; kt += 4){
    int kb = kt*16;
    const bf16* krow = Qb + (size_t)(batch*2048 + kb + ln)*512 + 8*kq;
    v8f acc = vzero();
#pragma unroll
    for(int kc=0; kc<16; ++kc){
      acc = wmma_bf(ldA(qrow + kc*32), ldB(krow + kc*32), acc);
    }
#pragma unroll
    for(int v=0; v<8; ++v) sc[(8*kq + v)*2048 + kb + ln] = acc[v]*scale;
  }
  __syncthreads();

  int row = tid >> 3, seg = tid & 7;
  const float* rp = sc + row*2048 + seg*256;
  float m = -3.0e38f;
  for(int i=0;i<256;++i) m = fmaxf(m, rp[i]);
  pmax[row*8 + seg] = m;
  __syncthreads();
  if(tid < 16){ float mm = pmax[tid*8]; for(int i=1;i<8;++i) mm = fmaxf(mm, pmax[tid*8+i]); rmax[tid] = mm; }
  __syncthreads();
  float rm = rmax[row];
  float s = 0.f;
  for(int i=0;i<256;++i) s += expf(rp[i] - rm);
  psum[row*8 + seg] = s;
  __syncthreads();
  if(tid < 16){ float ss = 0.f; for(int i=0;i<8;++i) ss += psum[tid*8+i]; rsum[tid] = 1.0f/ss; }
  __syncthreads();
  float inv = rsum[row];
  bf16* arow = A + (size_t)(qg + row)*2048 + seg*256;
  for(int i=0;i<256;++i) arow[i] = tob(expf(rp[i] - rm)*inv);
}

// ----------------------- K4: O = A @ V  (bf16 GEMM) ------------------------
__global__ __launch_bounds__(256) void k_av(const bf16* __restrict__ A,
                                            const bf16* __restrict__ Vb,
                                            bf16* __restrict__ O){
  int tid = threadIdx.x, lane = tid & 31;
  int ln = lane & 15, kq = lane >> 4;
  int wv = wave_id();
  int qbase = blockIdx.x*16;
  int batch = qbase >> 11;
  const bf16* arow  = A + (size_t)(qbase + ln)*2048 + 8*kq;
  const bf16* vbase = Vb + (size_t)batch*1024*2048 + 8*kq + (size_t)(wv*8)*16*2048;
  v8f acc[8];
#pragma unroll
  for(int c=0;c<8;++c) acc[c] = vzero();
  for(int kt=0; kt<64; ++kt){
    if(kt < 63) __builtin_prefetch(arow + (kt+1)*32, 0, 0);
    v16bf a = ldA(arow + kt*32);
#pragma unroll
    for(int c=0;c<8;++c){
      acc[c] = wmma_bf(a, ldB(vbase + (size_t)(c*16 + ln)*2048 + kt*32), acc[c]);
    }
  }
#pragma unroll
  for(int c=0;c<8;++c){
    int col = (wv*8 + c)*16 + ln;
#pragma unroll
    for(int v=0;v<8;++v) O[(size_t)(qbase + 8*kq + v)*1024 + col] = tob(acc[c][v]);
  }
}

// ------- K5: xb1 = x + equi_linear(w2, O); also emit normed bf16 xb1 -------
template<int J, int JJ>
DEV void k5_phase1(const bf16* inb, const bf16* __restrict__ Wp,
                   const float* __restrict__ x, float* __restrict__ xb1,
                   float* ssum, int tokbase, int ln, int kq, float vals[2][4][8]){
#pragma unroll
  for(int ct=0;ct<4;++ct){
    int o = ct*16 + ln;
    v8f acc = lin_tile<J,4096>(inb, Wp, o, ln, kq);
#pragma unroll
    for(int v=0;v<8;++v){
      int t = tokbase + 8*kq + v;
      float val = acc[v] + x[(size_t)t*1024 + o*16 + J];
      xb1[(size_t)t*1024 + J*64 + o] = val;
      vals[JJ][ct][v] = val;
      if constexpr (ISINNER[J]) atomicAdd(&ssum[8*kq + v], val*val);
    }
  }
}
template<int J, int JJ>
DEV void k5_phase2(bf16* __restrict__ nout, const float* ssum,
                   int tokbase, int ln, int kq, const float vals[2][4][8]){
#pragma unroll
  for(int ct=0;ct<4;++ct){
    int o = ct*16 + ln;
#pragma unroll
    for(int v=0;v<8;++v){
      int t = tokbase + 8*kq + v;
      nout[(size_t)t*1024 + J*64 + o] = tob(vals[JJ][ct][v]*ssum[8*kq + v]);
    }
  }
}

__global__ __launch_bounds__(256) void k_lin_res_norm(const bf16* __restrict__ in,
                                                      const bf16* __restrict__ Wp,
                                                      const float* __restrict__ x,
                                                      float* __restrict__ xb1,
                                                      bf16* __restrict__ nout){
  extern __shared__ char smem[];
  float* ssum = (float*)smem;   // [16]
  int tid = threadIdx.x, lane = tid & 31;
  int ln = lane & 15, kq = lane >> 4;
  int wv = wave_id();
  int tokbase = blockIdx.x*16;
  if(tid < 16) ssum[tid] = 0.f;
  __syncthreads();
  const bf16* inb = in + (size_t)tokbase*1024;
  float vals[2][4][8];
#define K5P1(W) { k5_phase1<2*W,0>(inb, Wp, x, xb1, ssum, tokbase, ln, kq, vals); \
                  k5_phase1<2*W+1,1>(inb, Wp, x, xb1, ssum, tokbase, ln, kq, vals); }
  WAVE8(wv, K5P1)
#undef K5P1
  __syncthreads();
  if(tid < 16) ssum[tid] = rsqrtf(ssum[tid]*(1.0f/64.0f) + 1e-6f);
  __syncthreads();
#define K5P2(W) { k5_phase2<2*W,0>(nout, ssum, tokbase, ln, kq, vals); \
                  k5_phase2<2*W+1,1>(nout, ssum, tokbase, ln, kq, vals); }
  WAVE8(wv, K5P2)
#undef K5P2
}

// ------------------ K6: plain equi_linear (bf16 -> bf16) -------------------
template<int J>
DEV void k6_comp(const bf16* inb, const bf16* __restrict__ Wp,
                 bf16* __restrict__ out, int tokbase, int ln, int kq){
#pragma unroll
  for(int ct=0;ct<4;++ct){
    int o = ct*16 + ln;
    v8f acc = lin_tile<J,4096>(inb, Wp, o, ln, kq);
#pragma unroll
    for(int v=0;v<8;++v)
      out[(size_t)(tokbase + 8*kq + v)*1024 + J*64 + o] = tob(acc[v]);
  }
}
__global__ __launch_bounds__(256) void k_lin(const bf16* __restrict__ in,
                                             const bf16* __restrict__ Wp,
                                             bf16* __restrict__ out){
  int tid = threadIdx.x, lane = tid & 31;
  int ln = lane & 15, kq = lane >> 4;
  int wv = wave_id();
  int tokbase = blockIdx.x*16;
  const bf16* inb = in + (size_t)tokbase*1024;
#define K6W(W) { k6_comp<2*W>(inb, Wp, out, tokbase, ln, kq); \
                 k6_comp<2*W+1>(inb, Wp, out, tokbase, ln, kq); }
  WAVE8(wv, K6W)
#undef K6W
}

// ------------- K7: half-width equi_linear (out=32 ch) -> f32 ---------------
template<int J>
DEV void k7_comp(const bf16* inb, const bf16* __restrict__ Wp,
                 float* __restrict__ out, int tokbase, int ln, int kq){
#pragma unroll
  for(int ct=0;ct<2;++ct){
    int o = ct*16 + ln;
    v8f acc = lin_tile<J,2048>(inb, Wp, o, ln, kq);
#pragma unroll
    for(int v=0;v<8;++v)
      out[(size_t)(tokbase + 8*kq + v)*512 + J*32 + o] = acc[v];
  }
}
__global__ __launch_bounds__(256) void k_lin_half(const bf16* __restrict__ in,
                                                  const bf16* __restrict__ Wp,
                                                  float* __restrict__ out){
  int tid = threadIdx.x, lane = tid & 31;
  int ln = lane & 15, kq = lane >> 4;
  int wv = wave_id();
  int tokbase = blockIdx.x*16;
  const bf16* inb = in + (size_t)tokbase*1024;
#define K7W(W) { k7_comp<2*W>(inb, Wp, out, tokbase, ln, kq); \
                 k7_comp<2*W+1>(inb, Wp, out, tokbase, ln, kq); }
  WAVE8(wv, K7W)
#undef K7W
}

// -------- K8: geometric bilinear (GP + equivariant join) + gated GELU ------
__global__ __launch_bounds__(256) void k_bilinear(const float* __restrict__ L,
                                                  const float* __restrict__ R,
                                                  const float* __restrict__ JL,
                                                  const float* __restrict__ JR,
                                                  const float* __restrict__ ref,
                                                  bf16* __restrict__ h){
  int gid = blockIdx.x*256 + threadIdx.x;     // 4096 tokens * 32 channels
  if(gid >= 4096*32) return;
  int t = gid >> 5, ch = gid & 31;
  float l[16], r[16], jl[16], jr[16];
  size_t base = (size_t)t*512 + ch;
#pragma unroll
  for(int j=0;j<16;++j){
    l[j]  = L[base + j*32];  r[j]  = R[base + j*32];
    jl[j] = JL[base + j*32]; jr[j] = JR[base + j*32];
  }
  float gp[16], jn[16];
#pragma unroll
  for(int j=0;j<16;++j){ gp[j] = 0.f; jn[j] = 0.f; }
#pragma unroll
  for(int a=0;a<16;++a){
#pragma unroll
    for(int b=0;b<16;++b){
      if constexpr (true){
        if(TB.gt[a][b] >= 0) gp[(int)TB.gt[a][b]] += (float)TB.gs[a][b]*l[a]*r[b];
        if(TB.jt[a][b] >= 0) jn[(int)TB.jt[a][b]] += (float)TB.js[a][b]*jl[a]*jr[b];
      }
    }
  }
  float r15 = ref[(t >> 11)*16 + 15];   // e0123 of reference, per batch
#pragma unroll
  for(int j=0;j<16;++j) jn[j] *= r15;
  float gg = gelu_exact(gp[0]);          // scalar-gated GELU per output channel
  float gj = gelu_exact(jn[0]);
#pragma unroll
  for(int j=0;j<16;++j){
    h[(size_t)t*1024 + j*64 + ch]      = tob(gp[j]*gg);
    h[(size_t)t*1024 + j*64 + 32 + ch] = tob(jn[j]*gj);
  }
}

// -------- K9: out = xb1 + equi_linear(w4, h), reference [b,s,c,k] layout ---
template<int J>
DEV void k9_comp(const bf16* inb, const bf16* __restrict__ Wp,
                 const float* __restrict__ xb1, float* __restrict__ out,
                 int tokbase, int ln, int kq){
#pragma unroll
  for(int ct=0;ct<4;++ct){
    int o = ct*16 + ln;
    v8f acc = lin_tile<J,4096>(inb, Wp, o, ln, kq);
#pragma unroll
    for(int v=0;v<8;++v){
      int t = tokbase + 8*kq + v;
      out[(size_t)t*1024 + o*16 + J] = acc[v] + xb1[(size_t)t*1024 + J*64 + o];
    }
  }
}
__global__ __launch_bounds__(256) void k_lin_res_out(const bf16* __restrict__ in,
                                                     const bf16* __restrict__ Wp,
                                                     const float* __restrict__ xb1,
                                                     float* __restrict__ out){
  int tid = threadIdx.x, lane = tid & 31;
  int ln = lane & 15, kq = lane >> 4;
  int wv = wave_id();
  int tokbase = blockIdx.x*16;
  const bf16* inb = in + (size_t)tokbase*1024;
#define K9W(W) { k9_comp<2*W>(inb, Wp, xb1, out, tokbase, ln, kq); \
                 k9_comp<2*W+1>(inb, Wp, xb1, out, tokbase, ln, kq); }
  WAVE8(wv, K9W)
#undef K9W
}

// ------------------------------ host launch --------------------------------
// Workspace layout (lifetime-overlapped regions; peak ~65MB):
constexpr size_t OFF_W1  = 0,      SZ_WFULL = 9*64*64*2;  // 73728 each
constexpr size_t OFF_W2  = OFF_W1 + SZ_WFULL;
constexpr size_t OFF_W3  = OFF_W2 + SZ_WFULL;
constexpr size_t OFF_W4  = OFF_W3 + SZ_WFULL;
constexpr size_t SZ_WHALF = 9*32*64*2;                    // 36864 each
constexpr size_t OFF_WL  = OFF_W4 + SZ_WFULL;
constexpr size_t OFF_WR  = OFF_WL + SZ_WHALF;
constexpr size_t OFF_WJL = OFF_WR + SZ_WHALF;
constexpr size_t OFF_WJR = OFF_WJL + SZ_WHALF;
constexpr size_t MB = 1ull << 20;
constexpr size_t OFF_XB1 = 1*MB;              // [4096][16][64] f32, K5..K9
constexpr size_t OFF_A   = OFF_XB1 + 16*MB;   // [4096][2048] bf16, K3..K4
constexpr size_t OFF_L   = OFF_A;             // reuse: [4096][16][32] f32, K7..K8
constexpr size_t OFF_R   = OFF_A + 8*MB;
constexpr size_t OFF_QB  = OFF_XB1 + 32*MB;   // [4096][512] bf16, K2..K3
constexpr size_t OFF_VB  = OFF_QB + 4*MB;     // [2][1024][2048] bf16, K2..K4
constexpr size_t OFF_JL  = OFF_QB;            // reuse after K4
constexpr size_t OFF_JR  = OFF_QB + 8*MB;
constexpr size_t OFF_O   = OFF_XB1 + 48*MB;   // [4096][1024] bf16, K4..K5
constexpr size_t OFF_H3  = OFF_O;             // reuse after K5 (K6..K7)
constexpr size_t OFF_H   = OFF_O;             // reuse after K7 (K8..K9)
constexpr size_t OFF_NX  = OFF_XB1 + 56*MB;   // [4096][16][64] bf16, K5..K6

extern "C" void kernel_launch(void* const* d_in, const int* in_sizes, int n_in,
                              void* d_out, int out_size, void* d_ws, size_t ws_size,
                              hipStream_t stream) {
  (void)in_sizes; (void)n_in; (void)out_size; (void)ws_size;
  const float* x   = (const float*)d_in[0];
  const float* ref = (const float*)d_in[1];
  const float* w1  = (const float*)d_in[2];
  const float* w2  = (const float*)d_in[3];
  const float* w3  = (const float*)d_in[4];
  const float* wl  = (const float*)d_in[5];
  const float* wr  = (const float*)d_in[6];
  const float* wjl = (const float*)d_in[7];
  const float* wjr = (const float*)d_in[8];
  const float* w4  = (const float*)d_in[9];

  char* ws = (char*)d_ws;
  bf16* Wp1 = (bf16*)(ws + OFF_W1);
  bf16* Wp2 = (bf16*)(ws + OFF_W2);
  bf16* Wp3 = (bf16*)(ws + OFF_W3);
  bf16* Wp4 = (bf16*)(ws + OFF_W4);
  bf16* Wpl = (bf16*)(ws + OFF_WL);
  bf16* Wpr = (bf16*)(ws + OFF_WR);
  bf16* Wpjl= (bf16*)(ws + OFF_WJL);
  bf16* Wpjr= (bf16*)(ws + OFF_WJR);
  bf16* Qb  = (bf16*)(ws + OFF_QB);
  bf16* Vb  = (bf16*)(ws + OFF_VB);
  bf16* Ab  = (bf16*)(ws + OFF_A);
  bf16* Ob  = (bf16*)(ws + OFF_O);
  float* xb1= (float*)(ws + OFF_XB1);
  bf16* nxb = (bf16*)(ws + OFF_NX);
  bf16* h3  = (bf16*)(ws + OFF_H3);
  float* Lb = (float*)(ws + OFF_L);
  float* Rb = (float*)(ws + OFF_R);
  float* JLb= (float*)(ws + OFF_JL);
  float* JRb= (float*)(ws + OFF_JR);
  bf16* hb  = (bf16*)(ws + OFF_H);

  // K1: transpose weights to [basis][out][in] bf16
  k_prep_w<<<(9*64*64 + 255)/256, 256, 0, stream>>>(w1, Wp1, 64);
  k_prep_w<<<(9*64*64 + 255)/256, 256, 0, stream>>>(w2, Wp2, 64);
  k_prep_w<<<(9*64*64 + 255)/256, 256, 0, stream>>>(w3, Wp3, 64);
  k_prep_w<<<(9*64*64 + 255)/256, 256, 0, stream>>>(w4, Wp4, 64);
  k_prep_w<<<(9*32*64 + 255)/256, 256, 0, stream>>>(wl,  Wpl, 32);
  k_prep_w<<<(9*32*64 + 255)/256, 256, 0, stream>>>(wr,  Wpr, 32);
  k_prep_w<<<(9*32*64 + 255)/256, 256, 0, stream>>>(wjl, Wpjl, 32);
  k_prep_w<<<(9*32*64 + 255)/256, 256, 0, stream>>>(wjr, Wpjr, 32);

  // K2: norm + equi_linear(w1) -> Q/V   (LDS: 64KB stage + 32KB normed + red)
  k_norm_lin1<<<256, 256, 65536 + 32768 + 64, stream>>>(x, Wp1, Qb, Vb);
  // K3: scores + softmax (LDS: 128KB score rows + reductions)
  k_scores<<<256, 128, 131072 + 512 + 512 + 64 + 64, stream>>>(Qb, Ab);
  // K4: O = A @ V
  k_av<<<256, 256, 0, stream>>>(Ab, Vb, Ob);
  // K5: xb1 = x + equi_linear(w2, O); normed bf16 copy for branch 2
  k_lin_res_norm<<<256, 256, 64, stream>>>(Ob, Wp2, x, xb1, nxb);
  // K6: h3 = equi_linear(w3, norm(xb1))
  k_lin<<<256, 256, 0, stream>>>(nxb, Wp3, h3);
  // K7: four half-width linears
  k_lin_half<<<256, 256, 0, stream>>>(h3, Wpl,  Lb);
  k_lin_half<<<256, 256, 0, stream>>>(h3, Wpr,  Rb);
  k_lin_half<<<256, 256, 0, stream>>>(h3, Wpjl, JLb);
  k_lin_half<<<256, 256, 0, stream>>>(h3, Wpjr, JRb);
  // K8: geometric product + join + scalar-gated GELU
  k_bilinear<<<512, 256, 0, stream>>>(Lb, Rb, JLb, JRb, ref, hb);
  // K9: out = xb1 + equi_linear(w4, h), written in reference [b,s,c,k] layout
  k_lin_res_out<<<256, 256, 0, stream>>>(hb, Wp4, xb1, (float*)d_out);
}
